// VectorQuantizerSTE_73735998538501
// MI455X (gfx1250) — matrix operand order repaired and checked by reading, not compiled, not run on hardware
//
#include <hip/hip_runtime.h>
#include <hip/hip_bf16.h>
#include <math.h>

#define B_ 8
#define S_ 2048
#define D_ 256
#define K_ 2048
#define N_ (B_ * S_)
#define BETA_ 0.25f
#define NORM_EPS_ 1e-12f

typedef __attribute__((ext_vector_type(16))) __bf16 v16bf;
typedef __attribute__((ext_vector_type(8))) float v8f;

union AFrag { v16bf v; uint4 u[2]; };
union Acc   { v8f v;  float f[8]; };

__device__ __forceinline__ unsigned short f32_to_bf16(float f) {
  unsigned x = __float_as_uint(f);
  x += 0x7FFFu + ((x >> 16) & 1u);   // round to nearest even
  return (unsigned short)(x >> 16);
}

// ---------------- K0: zero accumulators ----------------
__global__ void k_init(unsigned* __restrict__ hist, float* __restrict__ accum) {
  const int t = threadIdx.x;
  for (int k = t; k < K_; k += 256) hist[k] = 0u;
  if (t == 0) {
    accum[0] = 0.f;                          // sum (latent - q)^2
    accum[1] = 0.f;                          // sum selected cosine
    accum[2] = 0.f;                          // sum off-diag cdist
    ((unsigned*)accum)[3] = 0x7F800000u;     // min cdist bits (+inf)
  }
}

// ---------------- K1: repar = mean + std*codebook; norms; bf16 copy --------
__global__ void __launch_bounds__(256) k_repar(
    const float* __restrict__ codebook, const float* __restrict__ c_mean,
    const float* __restrict__ c_std, float* __restrict__ repar,
    unsigned short* __restrict__ reparb, float* __restrict__ sq_repar,
    float* __restrict__ norm_repar) {
  __shared__ float red[256];
  const int k = blockIdx.x, d = threadIdx.x;
  const float v = c_mean[d] + c_std[d] * codebook[k * D_ + d];
  repar[k * D_ + d]  = v;
  reparb[k * D_ + d] = f32_to_bf16(v);
  red[d] = v * v;
  __syncthreads();
  for (int s = 128; s > 0; s >>= 1) { if (d < s) red[d] += red[d + s]; __syncthreads(); }
  if (d == 0) { sq_repar[k] = red[0]; norm_repar[k] = sqrtf(red[0]); }
}

// ---------------- K2: flat norms + bf16 copy ----------------
__global__ void __launch_bounds__(256) k_flat(
    const float* __restrict__ latent, unsigned short* __restrict__ flatb,
    float* __restrict__ norm_flat) {
  __shared__ float red[256];
  const int r = blockIdx.x, d = threadIdx.x;
  const float v = latent[r * D_ + d];
  flatb[r * D_ + d] = f32_to_bf16(v);
  red[d] = v * v;
  __syncthreads();
  for (int s = 128; s > 0; s >>= 1) { if (d < s) red[d] += red[d + s]; __syncthreads(); }
  if (d == 0) norm_flat[r] = sqrtf(red[0]);
}

// ---------------- K3: fused WMMA distance-GEMM + argmax + gather -----------
__global__ void __launch_bounds__(256) k_main(
    const unsigned short* __restrict__ flatb,
    const unsigned short* __restrict__ reparb,
    const float* __restrict__ repar,
    const float* __restrict__ latent,
    const float* __restrict__ sq_repar,
    const float* __restrict__ norm_flat,
    const float* __restrict__ norm_repar,
    float* __restrict__ out_q, float* __restrict__ out_idx,
    unsigned* __restrict__ hist, float* __restrict__ accum) {
  __shared__ float dotT[16][132];
  __shared__ float pm[16][16];
  __shared__ float pd[16][16];
  __shared__ int   pi[16][16];
  __shared__ float bestM[16];
  __shared__ float bestD[16];
  __shared__ int   bestI[16];
  __shared__ float red[256];

  const int tid  = threadIdx.x;
  const int wave = tid >> 5;
  const int lane = tid & 31;
  const int h    = lane >> 4;
  const int mn   = lane & 15;
  const int rowBase = blockIdx.x * 16;
  const int row = rowBase + mn;

  // Preload A fragments: 16 rows x 256 K, kept in registers across col loop.
  AFrag a[8];
#pragma unroll
  for (int kc = 0; kc < 8; ++kc) {
    const int kb = kc * 32;
    const uint4* p0 = (const uint4*)(flatb + row * D_ + kb + 8 * h);
    const uint4* p1 = (const uint4*)(flatb + row * D_ + kb + 16 + 8 * h);
    a[kc].u[0] = p0[0];
    a[kc].u[1] = p1[0];
  }

  if (tid < 16) { bestM[tid] = -INFINITY; bestI[tid] = 0; bestD[tid] = 0.f; }
  __syncthreads();

  for (int ct = 0; ct < K_ / 128; ++ct) {
    const int colBase = ct * 128;
    const int colTile = colBase + wave * 16;
    Acc acc;
#pragma unroll
    for (int i = 0; i < 8; ++i) acc.f[i] = 0.f;
#pragma unroll
    for (int kc = 0; kc < 8; ++kc) {
      const int kb = kc * 32;
      AFrag b;
      const uint4* pb = (const uint4*)(reparb + (colTile + mn) * D_ + kb + 8 * h);
      b.u[0] = pb[0];
      b.u[1] = pb[2];
      acc.v = __builtin_amdgcn_wmma_f32_16x16x32_bf16(
          false, a[kc].v, false, b.v, (short)0, acc.v, false, false);
    }
#pragma unroll
    for (int v = 0; v < 8; ++v) dotT[v + 8 * h][wave * 16 + mn] = acc.f[v];
    __syncthreads();

    // per-row partial argmax of metric = 2*dot - |r|^2  (== argmax of logits)
    {
      const int r = tid & 15;
      const int g = tid >> 4;
      float bm = -INFINITY; int bi = 0; float bd = 0.f;
#pragma unroll
      for (int c0 = 0; c0 < 8; ++c0) {
        const int c = g * 8 + c0;
        const int j = colBase + c;
        const float dot = dotT[r][c];
        const float m = 2.0f * dot - sq_repar[j];
        if (m > bm) { bm = m; bi = j; bd = dot; }
      }
      pm[r][g] = bm; pi[r][g] = bi; pd[r][g] = bd;
    }
    __syncthreads();
    if (tid < 16) {
      float bm = bestM[tid]; int bi = bestI[tid]; float bd = bestD[tid];
#pragma unroll
      for (int g = 0; g < 16; ++g)
        if (pm[tid][g] > bm) { bm = pm[tid][g]; bi = pi[tid][g]; bd = pd[tid][g]; }
      bestM[tid] = bm; bestI[tid] = bi; bestD[tid] = bd;
    }
    __syncthreads();
  }

  // per-row epilogue: index, histogram, selected cosine
  if (tid < 16) {
    const int r = rowBase + tid;
    const int idx = bestI[tid];
    out_idx[r] = (float)idx;
    atomicAdd(&hist[idx], 1u);
    const float nf = fmaxf(norm_flat[r], NORM_EPS_);
    const float nr = fmaxf(norm_repar[idx], NORM_EPS_);
    atomicAdd(&accum[1], bestD[tid] / (nf * nr));
  }
  __syncthreads();

  // gather quantized rows (hard assign) + squared-diff accumulation
  float lsum = 0.f;
  for (int e = tid; e < 16 * D_; e += 256) {
    const int r = e >> 8;
    const int d = e & (D_ - 1);
    const int idx = bestI[r];
    const float q = repar[idx * D_ + d];
    const float l = latent[(rowBase + r) * D_ + d];
    out_q[(rowBase + r) * D_ + d] = q;
    const float df = l - q;
    lsum += df * df;
  }
  red[tid] = lsum;
  __syncthreads();
  for (int s = 128; s > 0; s >>= 1) { if (tid < s) red[tid] += red[tid + s]; __syncthreads(); }
  if (tid == 0) atomicAdd(&accum[0], red[0]);
}

// ---------------- K4: codebook diversity (repar @ repar^T via WMMA) --------
__global__ void __launch_bounds__(256) k_div(
    const unsigned short* __restrict__ reparb,
    const float* __restrict__ sq_repar,
    float* __restrict__ accum) {
  __shared__ float reds[256];
  __shared__ float redm[256];
  const int tid  = threadIdx.x;
  const int wave = tid >> 5;
  const int lane = tid & 31;
  const int h    = lane >> 4;
  const int mn   = lane & 15;
  const int iBlock = blockIdx.x >> 4;   // 0..127
  const int jBlock = blockIdx.x & 15;   // 0..15
  const int rowBase = iBlock * 16;
  const int colTile = jBlock * 128 + wave * 16;

  AFrag a[8];
#pragma unroll
  for (int kc = 0; kc < 8; ++kc) {
    const int kb = kc * 32;
    const uint4* p0 = (const uint4*)(reparb + (rowBase + mn) * D_ + kb + 8 * h);
    const uint4* p1 = (const uint4*)(reparb + (rowBase + mn) * D_ + kb + 16 + 8 * h);
    a[kc].u[0] = p0[0];
    a[kc].u[1] = p1[0];
  }
  Acc acc;
#pragma unroll
  for (int i = 0; i < 8; ++i) acc.f[i] = 0.f;
#pragma unroll
  for (int kc = 0; kc < 8; ++kc) {
    const int kb = kc * 32;
    AFrag b;
    const uint4* pb = (const uint4*)(reparb + (colTile + mn) * D_ + kb + 8 * h);
    b.u[0] = pb[0];
    b.u[1] = pb[2];
    acc.v = __builtin_amdgcn_wmma_f32_16x16x32_bf16(
        false, a[kc].v, false, b.v, (short)0, acc.v, false, false);
  }

  float ls = 0.f, lm = INFINITY;
#pragma unroll
  for (int v = 0; v < 8; ++v) {
    const int i = rowBase + v + 8 * h;
    const int j = colTile + mn;
    if (i != j) {
      const float cd2 = sq_repar[i] + sq_repar[j] - 2.0f * acc.f[v];
      const float c = sqrtf(fmaxf(cd2, 0.f));
      ls += c;
      lm = fminf(lm, c);
    }
  }
  reds[tid] = ls; redm[tid] = lm;
  __syncthreads();
  for (int s = 128; s > 0; s >>= 1) {
    if (tid < s) { reds[tid] += reds[tid + s]; redm[tid] = fminf(redm[tid], redm[tid + s]); }
    __syncthreads();
  }
  if (tid == 0) {
    atomicAdd(&accum[2], reds[0]);
    atomicMin((unsigned*)&accum[3], __float_as_uint(redm[0]));  // ok: values >= 0
  }
}

// ---------------- K5: finalize scalars ----------------
__global__ void __launch_bounds__(256) k_fin(
    const unsigned* __restrict__ hist, const float* __restrict__ accum,
    float* __restrict__ out_s) {
  __shared__ float red[256];
  const int t = threadIdx.x;
  float s = 0.f;
  for (int k = t; k < K_; k += 256) {
    const float p = (float)hist[k] / (float)N_;
    s += p * logf(p + 1e-10f);
  }
  red[t] = s;
  __syncthreads();
  for (int st = 128; st > 0; st >>= 1) { if (t < st) red[t] += red[t + st]; __syncthreads(); }
  if (t == 0) {
    const float sumSq = accum[0];
    const float mse = sumSq / (float)(N_ * D_);
    out_s[0] = BETA_ * mse;                                // commitment_loss
    out_s[1] = mse;                                        // codebook_loss
    out_s[2] = expf(-red[0]);                              // perplexity
    out_s[3] = accum[1] / (float)N_;                       // selected_cosine_sim
    out_s[4] = accum[2] / (float)((long long)K_ * (K_ - 1)); // avg_euclidean
    out_s[5] = __uint_as_float(((const unsigned*)accum)[3]); // min_euclidean
    out_s[6] = sqrtf(sumSq);                               // gradient_gap
  }
}

extern "C" void kernel_launch(void* const* d_in, const int* in_sizes, int n_in,
                              void* d_out, int out_size, void* d_ws, size_t ws_size,
                              hipStream_t stream) {
  (void)in_sizes; (void)n_in; (void)out_size; (void)ws_size;
  const float* latent   = (const float*)d_in[0];
  const float* codebook = (const float*)d_in[1];
  const float* c_mean   = (const float*)d_in[2];
  const float* c_std    = (const float*)d_in[3];

  char* ws = (char*)d_ws;
  size_t off = 0;
  float*          repar      = (float*)(ws + off);          off += (size_t)K_ * D_ * 4;
  unsigned short* reparb     = (unsigned short*)(ws + off); off += (size_t)K_ * D_ * 2;
  unsigned short* flatb      = (unsigned short*)(ws + off); off += (size_t)N_ * D_ * 2;
  float*          sq_repar   = (float*)(ws + off);          off += (size_t)K_ * 4;
  float*          norm_repar = (float*)(ws + off);          off += (size_t)K_ * 4;
  float*          norm_flat  = (float*)(ws + off);          off += (size_t)N_ * 4;
  unsigned*       hist       = (unsigned*)(ws + off);       off += (size_t)K_ * 4;
  float*          accum      = (float*)(ws + off);          off += 64;

  float* out     = (float*)d_out;
  float* out_q   = out;                       // [B,S,D]
  float* out_idx = out + (size_t)N_ * D_;     // [N] (as float)
  float* out_s   = out_idx + N_;              // 7 scalars

  k_init<<<1, 256, 0, stream>>>(hist, accum);
  k_repar<<<K_, 256, 0, stream>>>(codebook, c_mean, c_std, repar, reparb, sq_repar, norm_repar);
  k_flat<<<N_, 256, 0, stream>>>(latent, flatb, norm_flat);
  k_main<<<N_ / 16, 256, 0, stream>>>(flatb, reparb, repar, latent, sq_repar,
                                      norm_flat, norm_repar, out_q, out_idx, hist, accum);
  k_div<<<(K_ / 16) * (K_ / 128), 256, 0, stream>>>(reparb, sq_repar, accum);
  k_fin<<<1, 256, 0, stream>>>(hist, accum, out_s);
}
